// RaggedGravNet_38285338476785
// MI455X (gfx1250) — compile-verified, matrix operand order checked by reading, not run on hardware
//
#include <hip/hip_runtime.h>
#include <hip/hip_bf16.h>

// ---------- types for WMMA ----------
typedef __attribute__((ext_vector_type(16))) __bf16 v16bf;
typedef __attribute__((ext_vector_type(8)))  float  v8f;
typedef __attribute__((ext_vector_type(2)))  float  v2f;

#define FDIM   64
#define PDIM   64
#define NDIMS  4
#define NFILT  128
#define KNEI   63
#define KCAT   (2*PDIM + FDIM)   // 192

// k-index mapping for 16-bit A/B fragments of V_WMMA_*_16x16x32 (wave32).
// vgpr j (0..7), lane-half h (0/1), sub (0/1):
//   k = 16*(j>>2) + 8*h + 2*(j&3) + sub
__device__ __host__ __forceinline__ int kmap16(int j, int h, int sub) {
  return 16 * (j >> 2) + 8 * h + 2 * (j & 3) + sub;
}

__device__ __forceinline__ v16bf frag_a_f32(const float* src, int ld, int m, int h, int kk) {
  v16bf a;
#pragma unroll
  for (int e = 0; e < 16; ++e) {
    int j = e >> 1;
    int k = kk + kmap16(j, h, e & 1);
    a[e] = (__bf16)src[m * ld + k];
  }
  return a;
}

__device__ __forceinline__ v16bf frag_a_bf(const __bf16* src, int ld, int m, int h, int kk) {
  v16bf a;
#pragma unroll
  for (int e = 0; e < 16; ++e) {
    int j = e >> 1;
    int k = kk + kmap16(j, h, e & 1);
    a[e] = src[m * ld + k];
  }
  return a;
}

// ---------- kernel 0: prep ----------
// xbf: plain bf16 copy of x.
// wfeat_pk / wout_pk: B matrices pre-swizzled into per-lane WMMA fragment order:
//   packed[((ct*KC + kc)*32 + lane)*16 + e] = W[k][ct*16 + (lane&15)],
//   k = kc*32 + kmap16(e>>1, lane>>4, e&1)
// so a wave fetches a whole B fragment as two contiguous b128 loads.
__global__ void gravnet_prep(const float* __restrict__ x,
                             const float* __restrict__ w_feat,
                             const float* __restrict__ w_out,
                             __bf16* __restrict__ xbf,
                             __bf16* __restrict__ wfeat_pk,   // 4 ct * 2 kc * 32 * 16
                             __bf16* __restrict__ wout_pk,    // 8 ct * 6 kc * 32 * 16
                             int nx) {
  int i = blockIdx.x * blockDim.x + threadIdx.x;
  if (i < 4 * 2 * 32 * 16) {
    int e = i & 15, lane = (i >> 4) & 31, kc = (i >> 9) & 1, ct = i >> 10;
    int k = kc * 32 + kmap16(e >> 1, lane >> 4, e & 1);
    wfeat_pk[i] = (__bf16)w_feat[k * PDIM + ct * 16 + (lane & 15)];
  }
  if (i < 8 * 6 * 32 * 16) {
    int e = i & 15, lane = (i >> 4) & 31, kc = (i >> 9) % 6, ct = (i >> 9) / 6;
    int k = kc * 32 + kmap16(e >> 1, lane >> 4, e & 1);
    wout_pk[i] = (__bf16)w_out[k * NFILT + ct * 16 + (lane & 15)];
  }
  if (i < nx) xbf[i] = (__bf16)x[i];
}

// ---------- kernel 1: coords (VALU) + feat = relu(x@w_feat+b) (bf16 WMMA) ----------
__global__ void __launch_bounds__(128)
gravnet_transform(const float* __restrict__ x,
                  const __bf16* __restrict__ wfeat_pk,
                  const float* __restrict__ b_feat,
                  const float* __restrict__ w_spat,
                  float* __restrict__ feat,     // [N][64] ws
                  float* __restrict__ coords) { // [N][4]  -> d_out region
  __shared__ float xs[16][FDIM];
  const int tid = threadIdx.x;
  const int rowbase = blockIdx.x * 16;

  for (int i = tid; i < 16 * FDIM; i += blockDim.x) {
    xs[i / FDIM][i % FDIM] = x[(size_t)(rowbase + i / FDIM) * FDIM + (i % FDIM)];
  }
  __syncthreads();

  if (tid < 16 * NDIMS) {
    int r = tid >> 2, d = tid & 3;
    float s = 0.f;
#pragma unroll 8
    for (int k = 0; k < FDIM; ++k) s += xs[r][k] * w_spat[k * NDIMS + d];
    coords[(size_t)(rowbase + r) * NDIMS + d] = s;
  }

  const int wv = tid >> 5, lane = tid & 31;
  const int m = lane & 15, h = lane >> 4, n = m;
  const v16bf* bp = (const v16bf*)wfeat_pk;
  v8f acc = {};
#pragma unroll
  for (int kc = 0; kc < 2; ++kc) {
    v16bf a = frag_a_f32(&xs[0][0], FDIM, m, h, kc * 32);
    v16bf b = bp[(wv * 2 + kc) * 32 + lane];
    acc = __builtin_amdgcn_wmma_f32_16x16x32_bf16(false, a, false, b, (short)0, acc, false, false);
  }
#pragma unroll
  for (int v = 0; v < 8; ++v) {
    int row = v + 8 * h, col = 16 * wv + n;
    float val = acc[v] + b_feat[col];
    feat[(size_t)(rowbase + row) * PDIM + col] = fmaxf(val, 0.f);
  }
}

// ---------- kernel 2: distance Gram (f32 WMMA K=4) + top-63 selection ----------
__device__ __forceinline__ unsigned long long packkey(float v, int i) {
  unsigned u = __float_as_uint(v);
  u = (u & 0x80000000u) ? ~u : (u | 0x80000000u);
  return ((unsigned long long)u << 32) | (unsigned)i;
}

__global__ void __launch_bounds__(128)
gravnet_knn(const float* __restrict__ coords,  // [N][4]
            int* __restrict__ nidx,            // [N][63] -> d_out region
            float* __restrict__ dsq,           // [N][63] -> d_out region
            int S) {
  extern __shared__ float smem[];
  float* d2   = smem;                   // [16][S]
  float* qc   = smem + 16 * S;          // [16][4]
  float* sqq  = qc + 64;                // [16]
  float* cstg = sqq + 16;               // 4 waves * 2 bufs * 16 float4 = 512 floats

  const int tid = threadIdx.x, wv = tid >> 5, lane = tid & 31;
  const int tilesPerSeg = S / 16;
  const int seg  = blockIdx.x / tilesPerSeg;
  const int tile = blockIdx.x % tilesPerSeg;
  const int segbase = seg * S;
  const int qbase   = segbase + tile * 16;

  if (tid < 64) qc[tid] = coords[(size_t)(qbase + (tid >> 2)) * NDIMS + (tid & 3)];
  __syncthreads();
  if (tid < 16) {
    const float* q = &qc[tid * 4];
    sqq[tid] = q[0]*q[0] + q[1]*q[1] + q[2]*q[2] + q[3]*q[3];
  }
  __syncthreads();

  const int h = lane >> 4, mn = lane & 15;
  v2f afrag;
  afrag.x = qc[mn * 4 + 2 * h];
  afrag.y = qc[mn * 4 + 2 * h + 1];

  // phase 1: each wave computes d2 for S/4 candidates in 16-wide chunks.
  // Candidate float4 coords are staged into a per-wave LDS double buffer with
  // CDNA5 async copies (ASYNCcnt), overlapping transfer with WMMA + stores.
  float* mybuf = cstg + wv * 128;       // 2 x 16 float4
  const int cstart = wv * (S / 4), cend = cstart + (S / 4);

  {
    unsigned lds = (unsigned)(unsigned long long)&mybuf[0 * 64 + mn * 4];
    unsigned long long ga = (unsigned long long)(const void*)&coords[(size_t)(segbase + cstart + mn) * 4];
    asm volatile("global_load_async_to_lds_b128 %0, %1, off" :: "v"(lds), "v"(ga) : "memory");
  }
  int p = 0;
  for (int cb = cstart; cb < cend; cb += 16) {
    asm volatile("s_wait_asynccnt 0x0" ::: "memory");
    if (cb + 16 < cend) {
      unsigned lds = (unsigned)(unsigned long long)&mybuf[(p ^ 1) * 64 + mn * 4];
      unsigned long long ga = (unsigned long long)(const void*)&coords[(size_t)(segbase + cb + 16 + mn) * 4];
      asm volatile("global_load_async_to_lds_b128 %0, %1, off" :: "v"(lds), "v"(ga) : "memory");
    }
    float4 cc = *(const float4*)&mybuf[p * 64 + mn * 4];
    v2f bfrag;
    bfrag.x = h ? cc.z : cc.x;
    bfrag.y = h ? cc.w : cc.y;
    v8f c = {};
    c = __builtin_amdgcn_wmma_f32_16x16x4_f32(false, afrag, false, bfrag, (short)0, c, false, false);
    float sqc = cc.x*cc.x + cc.y*cc.y + cc.z*cc.z + cc.w*cc.w;
#pragma unroll
    for (int v = 0; v < 8; ++v) {
      int row = v + 8 * h;
      d2[row * S + cb + mn] = sqq[row] + sqc - 2.0f * c[v];
    }
    p ^= 1;
  }
  __syncthreads();

  // phase 2: each wave extracts the 63 smallest for 4 rows (ascending order)
  const float INF = __builtin_inff();
  for (int i = 0; i < 4; ++i) {
    const int r = 4 * wv + i;
    const int g = qbase + r;
    if (lane == 0) d2[r * S + tile * 16 + r] = INF;   // exclude self
    asm volatile("s_wait_dscnt 0" ::: "memory");
    for (int it = 0; it < KNEI; ++it) {
      unsigned long long best = packkey(INF, 0);
      for (int c = lane; c < S; c += 32) {
        unsigned long long k = packkey(d2[r * S + c], c);
        best = (k < best) ? k : best;
      }
#pragma unroll
      for (int off = 16; off > 0; off >>= 1) {
        unsigned long long o = __shfl_xor(best, off, 32);
        best = (o < best) ? o : best;
      }
      int idx = (int)(best & 0xffffffffu);
      unsigned b = (unsigned)(best >> 32);
      unsigned u = (b & 0x80000000u) ? (b ^ 0x80000000u) : ~b;
      float val = __uint_as_float(u);
      if (lane == 0) {
        nidx[(size_t)g * KNEI + it] = segbase + idx;
        dsq [(size_t)g * KNEI + it] = fmaxf(val, 0.f);
        d2[r * S + idx] = INF;
      }
      asm volatile("s_wait_dscnt 0" ::: "memory");
    }
  }
}

// ---------- kernel 3: gather/aggregate + out = tanh([agg,x]@w_out+b) ----------
__global__ void __launch_bounds__(128)
gravnet_out(const float* __restrict__ feat,     // [N][64]
            const __bf16* __restrict__ xbf,     // [N][64]
            const int* __restrict__ nidx,       // [N][63]
            const float* __restrict__ dsq,      // [N][63]
            const __bf16* __restrict__ wout_pk, // packed fragments
            const float* __restrict__ b_out,    // [128]
            float* __restrict__ out) {          // [N][128]
  __shared__ float  featT[16][PDIM];
  __shared__ __bf16 abuf[16][KCAT];   // [mean(64) | max(64) | x(64)] bf16

  const int tid = threadIdx.x, wv = tid >> 5, lane = tid & 31;
  const int rowbase = blockIdx.x * 16;

  for (int i = tid; i < 16 * PDIM; i += blockDim.x) {
    int r = i / PDIM, c = i % PDIM;
    featT[r][c] = feat[(size_t)(rowbase + r) * PDIM + c];
    abuf[r][2 * PDIM + c] = xbf[(size_t)(rowbase + r) * FDIM + c];
  }
  __syncthreads();

  // aggregation: each wave does 4 rows; each lane owns channels lane, lane+32
  for (int i = 0; i < 4; ++i) {
    const int r = 4 * wv + i;
    const int g = rowbase + r;
    const int f0 = lane, f1 = lane + 32;
    float sm0 = 0.f, sm1 = 0.f, mx0 = -__builtin_inff(), mx1 = -__builtin_inff();
    for (int k = 0; k < KNEI; ++k) {
      int   id = nidx[(size_t)g * KNEI + k];
      float w  = __expf(-10.0f * dsq[(size_t)g * KNEI + k]);
      const float* fr = feat + (size_t)id * PDIM;
      if (k + 1 < KNEI)
        __builtin_prefetch(feat + (size_t)nidx[(size_t)g * KNEI + k + 1] * PDIM, 0, 1);
      float v0 = w * fr[f0], v1 = w * fr[f1];
      sm0 += v0; sm1 += v1;
      mx0 = fmaxf(mx0, v0); mx1 = fmaxf(mx1, v1);
    }
    const float inv = 1.0f / (float)KNEI;
    abuf[r][f0]        = (__bf16)(sm0 * inv - featT[r][f0]);
    abuf[r][f1]        = (__bf16)(sm1 * inv - featT[r][f1]);
    abuf[r][PDIM + f0] = (__bf16)(mx0 - featT[r][f0]);
    abuf[r][PDIM + f1] = (__bf16)(mx1 - featT[r][f1]);
  }
  __syncthreads();

  // output GEMM: 16x192 @ 192x128, bf16 WMMA; 4 waves x 2 column tiles
  const int m = lane & 15, h = lane >> 4, n = m;
  const v16bf* bp = (const v16bf*)wout_pk;
#pragma unroll
  for (int t = 0; t < 2; ++t) {
    const int ct = 2 * wv + t;
    v8f acc = {};
#pragma unroll
    for (int kc = 0; kc < 6; ++kc) {
      v16bf a = frag_a_bf(&abuf[0][0], KCAT, m, h, kc * 32);
      v16bf b = bp[(ct * 6 + kc) * 32 + lane];
      acc = __builtin_amdgcn_wmma_f32_16x16x32_bf16(false, a, false, b, (short)0, acc, false, false);
    }
#pragma unroll
    for (int v = 0; v < 8; ++v) {
      int row = v + 8 * h, col = ct * 16 + n;
      out[(size_t)(rowbase + row) * NFILT + col] = tanhf(acc[v] + b_out[col]);
    }
  }
}

// ---------- host launcher ----------
extern "C" void kernel_launch(void* const* d_in, const int* in_sizes, int n_in,
                              void* d_out, int out_size, void* d_ws, size_t ws_size,
                              hipStream_t stream) {
  const float* x      = (const float*)d_in[0];
  const float* w_feat = (const float*)d_in[2];
  const float* b_feat = (const float*)d_in[3];
  const float* w_spat = (const float*)d_in[4];
  const float* w_out  = (const float*)d_in[5];
  const float* b_out  = (const float*)d_in[6];

  const int N = in_sizes[0] / FDIM;
  const int B = in_sizes[1] - 1;
  const int S = N / B;

  // d_out layout: out[N][128] | coords[N][4] | nidx[N][63] (int bits) | dsq[N][63]
  float* outp   = (float*)d_out;
  float* coords = outp + (size_t)N * NFILT;
  int*   nidxp  = (int*)(coords + (size_t)N * NDIMS);
  float* dsqp   = (float*)(nidxp + (size_t)N * KNEI);

  // workspace layout (all offsets 64B-aligned)
  char* w = (char*)d_ws;
  float*  feat     = (float*)w;                                     // N*64 f32
  __bf16* xbf      = (__bf16*)(w + (size_t)N * FDIM * 4);           // N*64 bf16
  __bf16* wfeat_pk = (__bf16*)(w + (size_t)N * FDIM * 4 + (size_t)N * FDIM * 2);
  __bf16* wout_pk  = wfeat_pk + 4 * 2 * 32 * 16;

  const int nx = N * FDIM;
  gravnet_prep<<<(nx + 255) / 256, 256, 0, stream>>>(x, w_feat, w_out, xbf, wfeat_pk, wout_pk, nx);

  gravnet_transform<<<N / 16, 128, 0, stream>>>(x, wfeat_pk, b_feat, w_spat, feat, coords);

  const size_t ldsB = (size_t)(16 * S + 64 + 16 + 512) * sizeof(float); // ~259 KB (CDNA5: 320 KB/WGP)
  gravnet_knn<<<N / 16, 128, ldsB, stream>>>(coords, nidxp, dsqp, S);

  gravnet_out<<<N / 16, 128, 0, stream>>>(feat, xbf, nidxp, dsqp, wout_pk, b_out, outp);
}
